// MSHGNN_65970697667196
// MI455X (gfx1250) — compile-verified, matrix operand order checked by reading, not run on hardware
//
#include <hip/hip_runtime.h>
#include <hip/hip_bf16.h>

#define NN 50000
#define EE 200000
#define DIN 64
#define HH 8
#define DD 64
#define HD 512
#define GG 500
#define RT 3125        // NN/16 row tiles
#define CTM 32         // HD/16 col tiles per matrix

typedef __attribute__((ext_vector_type(16))) __bf16 v16bf;
typedef __attribute__((ext_vector_type(8)))  float  v8f;

// ---------- helpers ----------
__device__ __forceinline__ unsigned short f2bf(float f) {
    unsigned u = __float_as_uint(f);
    u += 0x7FFFu + ((u >> 16) & 1u);   // round to nearest even
    return (unsigned short)(u >> 16);
}
// order-preserving float <-> uint encoding for atomic max
__device__ __forceinline__ unsigned fenc(float f) {
    unsigned u = __float_as_uint(f);
    return (u & 0x80000000u) ? ~u : (u | 0x80000000u);
}
__device__ __forceinline__ float fdec(unsigned e) {
    unsigned u = (e & 0x80000000u) ? (e & 0x7FFFFFFFu) : ~e;
    return __uint_as_float(u);
}
__device__ __forceinline__ void fadd(float* p, float v) {
    __hip_atomic_fetch_add(p, v, __ATOMIC_RELAXED, __HIP_MEMORY_SCOPE_AGENT);
}

// ---------- prep: swizzle weights into WMMA bf16 fragment layout ----------
// wswz[((mat*32+ct)*2+s)*32+lane][i], i in 0..15
__global__ void prep_weights(const float* W0, const float* W1, const float* W2, const float* W3,
                             const float* r0, const float* r1, const float* r2, const float* r3,
                             unsigned short* wswz) {
    int idx = blockIdx.x * blockDim.x + threadIdx.x;
    if (idx >= 5 * 2048) return;
    int lane = idx & 31, s = (idx >> 5) & 1, ct = (idx >> 6) & 31, mat = idx >> 11;
    int col = ct * 16 + (lane & 15);
    int kb  = s * 32 + ((lane & 16) ? 8 : 0);
    const float* W = (mat == 0) ? W0 : (mat == 1) ? W1 : (mat == 2) ? W2 : W3;
    #pragma unroll
    for (int i = 0; i < 16; ++i) {
        int K = kb + (i < 8 ? i : i + 8);
        float v;
        if (mat < 4) v = W[K * HD + col];
        else         v = r0[K * HD + col] + r1[K * HD + col] + r2[K * HD + col] + r3[K * HD + col];
        wswz[idx * 16 + i] = f2bf(v);
    }
}

__global__ void prep_bsum(const float* b0, const float* b1, const float* b2, const float* b3,
                          float* bsum) {
    int c = blockIdx.x * blockDim.x + threadIdx.x;
    if (c < HD) bsum[c] = b0[c] + b1[c] + b2[c] + b3[c];
}

// fswz[((rt*2+s)*32+lane)*16+i]
__global__ void prep_feat(const float* feat, unsigned short* fswz) {
    int idx = blockIdx.x * blockDim.x + threadIdx.x;
    if (idx >= RT * 64) return;
    int lane = idx & 31, s = (idx >> 5) & 1, rt = idx >> 6;
    int row = rt * 16 + (lane & 15);
    int kb  = s * 32 + ((lane & 16) ? 8 : 0);
    #pragma unroll
    for (int i = 0; i < 16; ++i) {
        int K = kb + (i < 8 ? i : i + 8);
        fswz[idx * 16 + i] = f2bf(feat[row * DIN + K]);
    }
}

// ---------- WMMA GEMM: [50000,64] x [64,512] -> fp32, optional bias ----------
__global__ void gemm_wmma(const unsigned short* __restrict__ aswz,
                          const unsigned short* __restrict__ bswz,
                          float* __restrict__ out, const float* __restrict__ bias) {
    int lane = threadIdx.x & 31, wave = threadIdx.x >> 5;
    int rt = blockIdx.x;
    int ct = blockIdx.y * 8 + wave;
    v16bf a0 = *(const v16bf*)(aswz + (rt * 64      + lane) * 16);
    v16bf a1 = *(const v16bf*)(aswz + (rt * 64 + 32 + lane) * 16);
    v16bf b0 = *(const v16bf*)(bswz + (ct * 64      + lane) * 16);
    v16bf b1 = *(const v16bf*)(bswz + (ct * 64 + 32 + lane) * 16);
    v8f c = {};
    c = __builtin_amdgcn_wmma_f32_16x16x32_bf16(false, a0, false, b0, (short)0, c, false, false);
    c = __builtin_amdgcn_wmma_f32_16x16x32_bf16(false, a1, false, b1, (short)0, c, false, false);
    int col = ct * 16 + (lane & 15);
    float bv = bias ? bias[col] : 0.0f;
    int rowbase = rt * 16 + ((lane >> 4) << 3);
    #pragma unroll
    for (int r = 0; r < 8; ++r)
        out[(size_t)(rowbase + r) * HD + col] = c[r] + bv;
}

// ---------- attention coefficients: el/er = sum_d h*al, h*ar ----------
__global__ void elr_kernel(const float* __restrict__ hbuf, const float* __restrict__ al,
                           const float* __restrict__ ar, float* __restrict__ el,
                           float* __restrict__ er) {
    int tid = blockIdx.x * blockDim.x + threadIdx.x;
    if (tid >= NN * HH) return;
    int n = tid >> 3, h = tid & 7;
    const float* hp = hbuf + (size_t)n * HD + h * DD;
    const float* ap = al + h * DD;
    const float* bp = ar + h * DD;
    float sl = 0.f, sr = 0.f;
    #pragma unroll 8
    for (int d = 0; d < DD; ++d) { float v = hp[d]; sl += v * ap[d]; sr += v * bp[d]; }
    el[tid] = sl; er[tid] = sr;
}

__global__ void init_ms(unsigned* menc, float* sden) {
    int tid = blockIdx.x * blockDim.x + threadIdx.x;
    if (tid < NN * HH) { menc[tid] = 0u; sden[tid] = 0.0f; }
}

// ---------- edge pass 1: leaky-relu score + atomic segment max ----------
__global__ void edge_max(const float* __restrict__ el, const float* __restrict__ er,
                         const int* __restrict__ src, const int* __restrict__ dst,
                         float* __restrict__ esc, unsigned* __restrict__ menc) {
    int tid = blockIdx.x * blockDim.x + threadIdx.x;
    if (tid >= EE * HH) return;
    int e = tid >> 3, h = tid & 7;
    int sN = src[e], dN = dst[e];
    float x = el[sN * HH + h] + er[dN * HH + h];
    x = x > 0.f ? x : 0.2f * x;
    esc[tid] = x;
    atomicMax(&menc[dN * HH + h], fenc(x));
}

// ---------- edge pass 2: exp + atomic segment sum ----------
__global__ void edge_exp(const int* __restrict__ dst, const unsigned* __restrict__ menc,
                         float* __restrict__ esc, float* __restrict__ sden) {
    int tid = blockIdx.x * blockDim.x + threadIdx.x;
    if (tid >= EE * HH) return;
    int e = tid >> 3, h = tid & 7;
    int dN = dst[e];
    float ex = __expf(esc[tid] - fdec(menc[dN * HH + h]));
    esc[tid] = ex;
    fadd(&sden[dN * HH + h], ex);
}

// ---------- edge pass 3: alpha * h[src] scattered into acc[dst] ----------
__global__ void edge_msg(const float* __restrict__ hbuf, const float* __restrict__ esc,
                         const float* __restrict__ sden, const int* __restrict__ src,
                         const int* __restrict__ dst, float* __restrict__ acc) {
    int e = blockIdx.x;
    int t = threadIdx.x;            // 256 threads, 2 floats each = 512 cols
    int sN = src[e], dN = dst[e];
    int h = t >> 5;                 // (t*2)/64
    float alpha = esc[e * HH + h] / sden[dN * HH + h];
    int col = t * 2;
    float2 hv = *(const float2*)(hbuf + (size_t)sN * HD + col);
    float* ap = acc + (size_t)dN * HD + col;
    fadd(ap,     alpha * hv.x);
    fadd(ap + 1, alpha * hv.y);
}

// ---------- segment mean of input features (seg = n/100) ----------
__global__ void seg_mean(const float* __restrict__ feat, float* __restrict__ meanb) {
    int g = blockIdx.x, d = threadIdx.x;     // 500 x 64
    float s = 0.f;
    for (int i = 0; i < 100; ++i) s += feat[(size_t)(g * 100 + i) * DIN + d];
    meanb[g * DIN + d] = s * 0.01f;
}

// ---------- final: mean[seg] + max over heads ----------
__global__ void final_out(const float* __restrict__ acc, const float* __restrict__ meanb,
                          float* __restrict__ out) {
    int tid = blockIdx.x * blockDim.x + threadIdx.x;
    if (tid >= NN * DD) return;
    int n = tid >> 6, d = tid & 63;
    float mx = -3.402823466e+38f;
    #pragma unroll
    for (int h = 0; h < HH; ++h) mx = fmaxf(mx, acc[(size_t)n * HD + h * DD + d]);
    out[tid] = meanb[(n / 100) * DIN + d] + mx;
}

extern "C" void kernel_launch(void* const* d_in, const int* in_sizes, int n_in,
                              void* d_out, int out_size, void* d_ws, size_t ws_size,
                              hipStream_t stream) {
    (void)in_sizes; (void)n_in; (void)out_size; (void)ws_size;
    const float* feat = (const float*)d_in[0];
    const float* W[4]  = {(const float*)d_in[1],  (const float*)d_in[6],
                          (const float*)d_in[11], (const float*)d_in[16]};
    const float* al[4] = {(const float*)d_in[2],  (const float*)d_in[7],
                          (const float*)d_in[12], (const float*)d_in[17]};
    const float* ar[4] = {(const float*)d_in[3],  (const float*)d_in[8],
                          (const float*)d_in[13], (const float*)d_in[18]};
    const float* bb[4] = {(const float*)d_in[4],  (const float*)d_in[9],
                          (const float*)d_in[14], (const float*)d_in[19]};
    const float* rW[4] = {(const float*)d_in[5],  (const float*)d_in[10],
                          (const float*)d_in[15], (const float*)d_in[20]};
    const int* src_a = (const int*)d_in[21];
    const int* dst_a = (const int*)d_in[22];
    const int* src_e = (const int*)d_in[23];
    const int* dst_e = (const int*)d_in[24];
    // conv2 runs on the reversed graph: swap src/dst
    const int* srcs[4] = {src_a, src_e, dst_a, dst_e};
    const int* dsts[4] = {dst_a, dst_e, src_a, src_e};

    char* wsb = (char*)d_ws;
    float*          acc   = (float*)         (wsb + 0);            // N*512 f32
    float*          hbuf  = (float*)         (wsb + 102400000ull); // N*512 f32
    unsigned short* fswz  = (unsigned short*)(wsb + 204800000ull); // N*64 bf16
    unsigned short* wswz  = (unsigned short*)(wsb + 211200000ull); // 5*2048*16 bf16
    float*          bsum  = (float*)         (wsb + 211527680ull); // 512
    float*          el    = (float*)         (wsb + 211529728ull); // N*8
    float*          er    = (float*)         (wsb + 213129728ull); // N*8
    unsigned*       menc  = (unsigned*)      (wsb + 214729728ull); // N*8
    float*          sden  = (float*)         (wsb + 216329728ull); // N*8
    float*          esc   = (float*)         (wsb + 217929728ull); // E*8
    float*          meanb = (float*)         (wsb + 224329728ull); // 500*64
    float*          out   = (float*)d_out;

    // prep
    prep_weights<<<(5 * 2048 + 255) / 256, 256, 0, stream>>>(
        W[0], W[1], W[2], W[3], rW[0], rW[1], rW[2], rW[3], wswz);
    prep_bsum<<<2, 256, 0, stream>>>(bb[0], bb[1], bb[2], bb[3], bsum);
    prep_feat<<<(RT * 64 + 255) / 256, 256, 0, stream>>>(feat, fswz);

    // acc = feat @ (sum rW) + (sum b)   (WMMA)
    gemm_wmma<<<dim3(RT, 4), 256, 0, stream>>>(fswz, wswz + 4 * 2048 * 16, acc, bsum);

    for (int k = 0; k < 4; ++k) {
        // h = feat @ W_k   (WMMA)
        gemm_wmma<<<dim3(RT, 4), 256, 0, stream>>>(fswz, wswz + k * 2048 * 16, hbuf,
                                                   (const float*)nullptr);
        elr_kernel<<<(NN * HH + 255) / 256, 256, 0, stream>>>(hbuf, al[k], ar[k], el, er);
        init_ms<<<(NN * HH + 255) / 256, 256, 0, stream>>>(menc, sden);
        edge_max<<<(EE * HH + 255) / 256, 256, 0, stream>>>(el, er, srcs[k], dsts[k], esc, menc);
        edge_exp<<<(EE * HH + 255) / 256, 256, 0, stream>>>(dsts[k], menc, esc, sden);
        edge_msg<<<EE, 256, 0, stream>>>(hbuf, esc, sden, srcs[k], dsts[k], acc);
    }

    seg_mean<<<GG, 64, 0, stream>>>(feat, meanb);
    final_out<<<(NN * DD + 255) / 256, 256, 0, stream>>>(acc, meanb, out);
}
